// SelfAttention_74646531604865
// MI455X (gfx1250) — matrix operand order repaired and checked by reading, not compiled, not run on hardware
//
#include <hip/hip_runtime.h>
#include <hip/hip_bf16.h>
#include <math.h>

typedef _Float16 half_t;
typedef __attribute__((ext_vector_type(16))) _Float16 v16h;
typedef __attribute__((ext_vector_type(8)))  _Float16 v8h;
typedef __attribute__((ext_vector_type(4)))  _Float16 v4h;
typedef __attribute__((ext_vector_type(8)))  float    v8f;
typedef __attribute__((ext_vector_type(4)))  float    v4f;

#define DIMV 1024
#define NH   16
#define HD   64

// ---------------------------------------------------------------------------
// Fragment loaders (CDNA5 WMMA 16x16x32 f16 layouts, wave32).
//  A 16x32 (MxK): lane m=lane&15; lanes 0-15 hold K 0-7 & 16-23,
//                 lanes 16-31 hold K 8-15 & 24-31 (2 halves/VGPR).
//  B 32x16 (KxN): lane n=lane&15; lanes 0-15 hold K 0-7 & 16-23,
//                 lanes 16-31 hold K 8-15 & 24-31 (same as A).
// Both sources are "NT": element (i,k) at base[i*ld + k], k contiguous.
// ---------------------------------------------------------------------------
__device__ __forceinline__ v16h load_a_frag(const half_t* base, int ld, int lane) {
  const int m  = lane & 15;
  const int hi = lane >> 4;
  const half_t* p = base + (size_t)m * ld;
  v8h c0 = *reinterpret_cast<const v8h*>(p + hi * 8);
  v8h c1 = *reinterpret_cast<const v8h*>(p + 16 + hi * 8);
  v16h r;
#pragma unroll
  for (int t = 0; t < 8; ++t) { r[t] = c0[t]; r[t + 8] = c1[t]; }
  return r;
}

__device__ __forceinline__ v16h load_b_frag(const half_t* base, int ld, int lane) {
  const int n  = lane & 15;
  const int hi = lane >> 4;
  const half_t* p = base + (size_t)n * ld + hi * 8;
  v8h c0 = *reinterpret_cast<const v8h*>(p);
  v8h c1 = *reinterpret_cast<const v8h*>(p + 16);
  v16h r;
#pragma unroll
  for (int t = 0; t < 8; ++t) { r[t] = c0[t]; r[t + 8] = c1[t]; }
  return r;
}

__device__ __forceinline__ v8f wmma_f16(v16h a, v16h b, v8f c) {
  return __builtin_amdgcn_wmma_f32_16x16x32_f16(false, a, false, b,
                                                (short)0, c, false, false);
}

// ---------------------------------------------------------------------------
// fp32 -> fp16 convert (x4 vectorized), optional nan_to_num + clip(+-10000)
// ---------------------------------------------------------------------------
__global__ __launch_bounds__(256) void cvt_f16_kernel(const float* __restrict__ in,
                                                      half_t* __restrict__ out,
                                                      int n4, int do_clip) {
  int i = blockIdx.x * blockDim.x + threadIdx.x;
  if (i >= n4) return;
  v4f v = *reinterpret_cast<const v4f*>(in + 4 * (size_t)i);
  v4h o;
#pragma unroll
  for (int t = 0; t < 4; ++t) {
    float f = v[t];
    if (!(fabsf(f) < 1e30f)) f = 0.f;            // NaN / inf -> 0
    if (do_clip) f = fminf(fmaxf(f, -10000.f), 10000.f);
    o[t] = (half_t)f;
  }
  *reinterpret_cast<v4h*>(out + 4 * (size_t)i) = o;
}

// ---------------------------------------------------------------------------
// C[M,N] = A[M,K] @ B[N,K]^T (+bias), f16 in, f32 out.
// One wave computes a 32x64 tile (2 A-frags x 4 B-frags = 8 WMMAs / k-step).
// k-loop unrolled x2 with ping-pong fragment sets: set X computes while set Y
// loads -- no rotation copies, no WAR movs into live WMMA sources.
// Requires K % 64 == 0.
// ---------------------------------------------------------------------------
__global__ __launch_bounds__(256) void gemm_nt_wmma(const half_t* __restrict__ A,
                                                    const half_t* __restrict__ B,
                                                    const float* __restrict__ bias,
                                                    float* __restrict__ C,
                                                    int M, int N, int K) {
  const int lane = threadIdx.x & 31;
  const int gw   = (blockIdx.x * blockDim.x + threadIdx.x) >> 5;
  const int ntiles = N >> 6;
  const int mt = gw / ntiles;                    // 32-row tile index
  const int nt = gw % ntiles;                    // 64-col tile index
  if (mt >= (M >> 5)) return;                    // wave-uniform

  const half_t* A0 = A + (size_t)(mt << 5) * K;
  const half_t* A1 = A0 + (size_t)16 * K;
  const half_t* B0 = B + (size_t)(nt << 6) * K;

  v8f acc[2][4];
#pragma unroll
  for (int h = 0; h < 2; ++h)
#pragma unroll
    for (int nb = 0; nb < 4; ++nb) acc[h][nb] = v8f{};

  v16h xa0, xa1, xb[4];   // ping set
  v16h ya0, ya1, yb[4];   // pong set

  // prologue: ping set holds k = 0
  xa0 = load_a_frag(A0, K, lane);
  xa1 = load_a_frag(A1, K, lane);
#pragma unroll
  for (int nb = 0; nb < 4; ++nb)
    xb[nb] = load_b_frag(B0 + ((size_t)nb << 4) * K, K, lane);

  int k0 = 0;
  for (; k0 + 64 < K; k0 += 64) {
    // load pong set @ k0+32 while computing ping set @ k0
    ya0 = load_a_frag(A0 + k0 + 32, K, lane);
    ya1 = load_a_frag(A1 + k0 + 32, K, lane);
#pragma unroll
    for (int nb = 0; nb < 4; ++nb)
      yb[nb] = load_b_frag(B0 + ((size_t)nb << 4) * K + k0 + 32, K, lane);
#pragma unroll
    for (int nb = 0; nb < 4; ++nb) {
      acc[0][nb] = wmma_f16(xa0, xb[nb], acc[0][nb]);
      acc[1][nb] = wmma_f16(xa1, xb[nb], acc[1][nb]);
    }
    // load ping set @ k0+64 while computing pong set @ k0+32
    xa0 = load_a_frag(A0 + k0 + 64, K, lane);
    xa1 = load_a_frag(A1 + k0 + 64, K, lane);
#pragma unroll
    for (int nb = 0; nb < 4; ++nb)
      xb[nb] = load_b_frag(B0 + ((size_t)nb << 4) * K + k0 + 64, K, lane);
#pragma unroll
    for (int nb = 0; nb < 4; ++nb) {
      acc[0][nb] = wmma_f16(ya0, yb[nb], acc[0][nb]);
      acc[1][nb] = wmma_f16(ya1, yb[nb], acc[1][nb]);
    }
  }
  // tail: ping holds k0, load pong @ k0+32, compute both (k0 == K-64)
  ya0 = load_a_frag(A0 + k0 + 32, K, lane);
  ya1 = load_a_frag(A1 + k0 + 32, K, lane);
#pragma unroll
  for (int nb = 0; nb < 4; ++nb)
    yb[nb] = load_b_frag(B0 + ((size_t)nb << 4) * K + k0 + 32, K, lane);
#pragma unroll
  for (int nb = 0; nb < 4; ++nb) {
    acc[0][nb] = wmma_f16(xa0, xb[nb], acc[0][nb]);
    acc[1][nb] = wmma_f16(xa1, xb[nb], acc[1][nb]);
  }
#pragma unroll
  for (int nb = 0; nb < 4; ++nb) {
    acc[0][nb] = wmma_f16(ya0, yb[nb], acc[0][nb]);
    acc[1][nb] = wmma_f16(ya1, yb[nb], acc[1][nb]);
  }

  const int n  = lane & 15;
  const int mb = (lane >> 4) << 3;
#pragma unroll
  for (int h = 0; h < 2; ++h)
#pragma unroll
    for (int nb = 0; nb < 4; ++nb) {
      int col = (nt << 6) + (nb << 4) + n;
      float badd = bias ? bias[col] : 0.f;
#pragma unroll
      for (int r = 0; r < 8; ++r)
        C[(size_t)((mt << 5) + (h << 4) + mb + r) * N + col] = acc[h][nb][r] + badd;
    }
}

// ---------------------------------------------------------------------------
// Per (b,h,l): nan_to_num, RMSNorm(q,k) in fp32, RoPE, emit f16:
//   q_h,k_h: [B*H, L, 64]   v_t: [B*H, 64, L] (transposed for P@V B-frags)
// ---------------------------------------------------------------------------
__global__ __launch_bounds__(256) void qkv_prep_kernel(const float* __restrict__ qkv,
                                                       const float* __restrict__ pe,
                                                       const float* __restrict__ qs,
                                                       const float* __restrict__ ks,
                                                       half_t* __restrict__ qh,
                                                       half_t* __restrict__ kh,
                                                       half_t* __restrict__ vt,
                                                       int L) {
  int tid = blockIdx.x * blockDim.x + threadIdx.x;
  int bh = tid / L;
  int l  = tid % L;
  int b = bh >> 4, h = bh & 15;

  const float* row = qkv + (size_t)(b * L + l) * (3 * DIMV);
  const float* qp = row + h * HD;
  const float* kp = row + DIMV + h * HD;
  const float* vp = row + 2 * DIMV + h * HD;

  float sq = 0.f, sk = 0.f;
#pragma unroll 8
  for (int d = 0; d < HD; ++d) {
    float q = qp[d]; if (!(fabsf(q) < 1e30f)) q = 0.f;
    float k = kp[d]; if (!(fabsf(k) < 1e30f)) k = 0.f;
    sq += q * q; sk += k * k;
  }
  float rq = rsqrtf(sq * (1.f / HD) + 1e-6f);
  float rk = rsqrtf(sk * (1.f / HD) + 1e-6f);

  const float* pel = pe + (size_t)l * (HD / 2) * 4;   // [l][i][c][e]
  half_t* qo = qh + ((size_t)bh * L + l) * HD;
  half_t* ko = kh + ((size_t)bh * L + l) * HD;
#pragma unroll 4
  for (int i = 0; i < HD / 2; ++i) {
    float q0 = qp[2 * i];     if (!(fabsf(q0) < 1e30f)) q0 = 0.f;
    float q1 = qp[2 * i + 1]; if (!(fabsf(q1) < 1e30f)) q1 = 0.f;
    float k0 = kp[2 * i];     if (!(fabsf(k0) < 1e30f)) k0 = 0.f;
    float k1 = kp[2 * i + 1]; if (!(fabsf(k1) < 1e30f)) k1 = 0.f;
    q0 *= rq * qs[2 * i]; q1 *= rq * qs[2 * i + 1];
    k0 *= rk * ks[2 * i]; k1 *= rk * ks[2 * i + 1];
    float p00 = pel[4 * i + 0], p01 = pel[4 * i + 1];
    float p10 = pel[4 * i + 2], p11 = pel[4 * i + 3];
    qo[2 * i]     = (half_t)(p00 * q0 + p01 * q1);
    qo[2 * i + 1] = (half_t)(p10 * q0 + p11 * q1);
    ko[2 * i]     = (half_t)(p00 * k0 + p01 * k1);
    ko[2 * i + 1] = (half_t)(p10 * k0 + p11 * k1);
  }
  half_t* vo = vt + (size_t)bh * HD * L;
#pragma unroll 8
  for (int d = 0; d < HD; ++d) {
    float v = vp[d]; if (!(fabsf(v) < 1e30f)) v = 0.f;
    vo[(size_t)d * L + l] = (half_t)v;
  }
}

// ---------------------------------------------------------------------------
// Flash attention: one wave per (b,h, 16 query rows), k-blocks of 32.
// All k/v fragments for the block are issued up front; the softmax VALU chain
// hides the v-load latency. P transposed C->A layout through LDS.
// Output f16 [B, L, DIM], clipped to +-10000.
// ---------------------------------------------------------------------------
__global__ __launch_bounds__(256) void flash_attn_kernel(const half_t* __restrict__ qh,
                                                         const half_t* __restrict__ kh,
                                                         const half_t* __restrict__ vt,
                                                         half_t* __restrict__ ao,
                                                         int L) {
  __shared__ half_t psh[8][16 * 32];
  const int lane = threadIdx.x & 31;
  const int wave = threadIdx.x >> 5;
  const int gw = blockIdx.x * 8 + wave;
  const int mtiles = L >> 4;
  const int bh = gw / mtiles;
  const int mt = gw % mtiles;
  const int b = bh >> 4, h = bh & 15;

  const half_t* qbase = qh + ((size_t)bh * L + (mt << 4)) * HD;
  const half_t* kbase = kh + (size_t)bh * L * HD;
  const half_t* vbase = vt + (size_t)bh * HD * L;
  half_t* pl = &psh[wave][0];

  v16h qf0 = load_a_frag(qbase, HD, lane);
  v16h qf1 = load_a_frag(qbase + 32, HD, lane);

  float mi[8], li[8];
  v8f acc[4] = { v8f{}, v8f{}, v8f{}, v8f{} };
#pragma unroll
  for (int r = 0; r < 8; ++r) { mi[r] = -1e30f; li[r] = 0.f; }

  const float sm_scale = 0.125f;   // 1/sqrt(64)
  const int n  = lane & 15;
  const int mb = (lane >> 4) << 3;

  for (int j0 = 0; j0 < L; j0 += 32) {
    // ---- issue ALL loads for this block first (k frags, then v frags)
    const half_t* kb0 = kbase + (size_t)j0 * HD;
    v16h kf[4];
    kf[0] = load_b_frag(kb0, HD, lane);            // keys j0..j0+15, d 0..31
    kf[1] = load_b_frag(kb0 + 32, HD, lane);       // keys j0..j0+15, d 32..63
    kf[2] = load_b_frag(kb0 + 16 * HD, HD, lane);  // keys j0+16..31, d 0..31
    kf[3] = load_b_frag(kb0 + 16 * HD + 32, HD, lane);
    v16h vf[4];
#pragma unroll
    for (int db = 0; db < 4; ++db)
      vf[db] = load_b_frag(vbase + ((size_t)db << 4) * L + j0, L, lane);

    // ---- S = q @ k^T for 16x32 tile (waits only on k-loads)
    v8f s[2];
    s[0] = wmma_f16(qf0, kf[0], v8f{});
    s[0] = wmma_f16(qf1, kf[1], s[0]);
    s[1] = wmma_f16(qf0, kf[2], v8f{});
    s[1] = wmma_f16(qf1, kf[3], s[1]);
#pragma unroll
    for (int nb = 0; nb < 2; ++nb)
#pragma unroll
      for (int r = 0; r < 8; ++r) s[nb][r] *= sm_scale;

    // ---- row max across 32 cols (cols live across the 16-lane half)
    float rmax[8];
#pragma unroll
    for (int r = 0; r < 8; ++r) rmax[r] = fmaxf(s[0][r], s[1][r]);
#pragma unroll
    for (int st = 1; st <= 8; st <<= 1)
#pragma unroll
      for (int r = 0; r < 8; ++r)
        rmax[r] = fmaxf(rmax[r], __shfl_xor(rmax[r], st, 32));

    float corr[8];
#pragma unroll
    for (int r = 0; r < 8; ++r) {
      float mn = fmaxf(mi[r], rmax[r]);
      corr[r] = __expf(mi[r] - mn);
      mi[r] = mn;
    }
    // ---- P = exp(S - m), row sum
    float rsum[8];
#pragma unroll
    for (int r = 0; r < 8; ++r) {
      s[0][r] = __expf(s[0][r] - mi[r]);
      s[1][r] = __expf(s[1][r] - mi[r]);
      rsum[r] = s[0][r] + s[1][r];
    }
#pragma unroll
    for (int st = 1; st <= 8; st <<= 1)
#pragma unroll
      for (int r = 0; r < 8; ++r)
        rsum[r] += __shfl_xor(rsum[r], st, 32);
#pragma unroll
    for (int r = 0; r < 8; ++r) li[r] = li[r] * corr[r] + rsum[r];

    // ---- rescale accumulators
#pragma unroll
    for (int db = 0; db < 4; ++db)
#pragma unroll
      for (int r = 0; r < 8; ++r) acc[db][r] *= corr[r];

    // ---- C-layout -> A-layout transpose of P through LDS (per-wave 1KB)
#pragma unroll
    for (int nb = 0; nb < 2; ++nb)
#pragma unroll
      for (int r = 0; r < 8; ++r)
        pl[(mb + r) * 32 + (nb << 4) + n] = (half_t)s[nb][r];
    __builtin_amdgcn_wave_barrier();          // keep DS store/load order
    v16h pf = load_a_frag(pl, 32, lane);
    __builtin_amdgcn_wave_barrier();

    // ---- out += P @ V  (v frags already resident)
#pragma unroll
    for (int db = 0; db < 4; ++db)
      acc[db] = wmma_f16(pf, vf[db], acc[db]);
  }

  // ---- epilogue: divide by l, clip, store f16 [B, L, DIM]
#pragma unroll
  for (int db = 0; db < 4; ++db)
#pragma unroll
    for (int r = 0; r < 8; ++r) {
      float o = acc[db][r] / li[r];
      o = fminf(fmaxf(o, -10000.f), 10000.f);
      ao[((size_t)b * L + (mt << 4) + mb + r) * DIMV + h * HD + (db << 4) + n] =
          (half_t)o;
    }
}

// ---------------------------------------------------------------------------
extern "C" void kernel_launch(void* const* d_in, const int* in_sizes, int n_in,
                              void* d_out, int out_size, void* d_ws, size_t ws_size,
                              hipStream_t stream) {
  const float* x       = (const float*)d_in[0];
  const float* pe      = (const float*)d_in[1];
  const float* qkv_w   = (const float*)d_in[2];
  const float* q_scale = (const float*)d_in[3];
  const float* k_scale = (const float*)d_in[4];
  const float* proj_w  = (const float*)d_in[5];
  const float* proj_b  = (const float*)d_in[6];
  float* out = (float*)d_out;

  const int L = in_sizes[1] / 128;         // pe: L * 32 * 2 * 2
  const int B = in_sizes[0] / (L * DIMV);
  const int M = B * L;                     // GEMM rows

  char* ws = (char*)d_ws;
  size_t off = 0;
  auto alloc = [&](size_t bytes) -> void* {
    off = (off + 255) & ~(size_t)255;
    void* p = ws + off;
    off += bytes;
    return p;
  };
  half_t* x_h   = (half_t*)alloc((size_t)M * DIMV * 2);
  half_t* wq_h  = (half_t*)alloc((size_t)3 * DIMV * DIMV * 2);
  half_t* wp_h  = (half_t*)alloc((size_t)DIMV * DIMV * 2);
  float*  qkv_f = (float*) alloc((size_t)M * 3 * DIMV * 4);
  half_t* q_h   = (half_t*)alloc((size_t)M * DIMV * 2);
  half_t* k_h   = (half_t*)alloc((size_t)M * DIMV * 2);
  half_t* v_t   = (half_t*)alloc((size_t)M * DIMV * 2);
  half_t* a_h   = (half_t*)alloc((size_t)M * DIMV * 2);

  // 1) converts (x4 vectorized)
  {
    int n4 = (M * DIMV) / 4;
    cvt_f16_kernel<<<(n4 + 255) / 256, 256, 0, stream>>>(x, x_h, n4, 1);
    n4 = (3 * DIMV * DIMV) / 4;
    cvt_f16_kernel<<<(n4 + 255) / 256, 256, 0, stream>>>(qkv_w, wq_h, n4, 0);
    n4 = (DIMV * DIMV) / 4;
    cvt_f16_kernel<<<(n4 + 255) / 256, 256, 0, stream>>>(proj_w, wp_h, n4, 0);
  }
  // 2) QKV GEMM: [M,1024] @ [3072,1024]^T -> f32
  {
    int waves = (M >> 5) * ((3 * DIMV) >> 6);
    gemm_nt_wmma<<<(waves * 32) / 256, 256, 0, stream>>>(
        x_h, wq_h, nullptr, qkv_f, M, 3 * DIMV, DIMV);
  }
  // 3) RMSNorm + RoPE + v-transpose
  {
    int threads = B * NH * L;
    qkv_prep_kernel<<<threads / 256, 256, 0, stream>>>(
        qkv_f, pe, q_scale, k_scale, q_h, k_h, v_t, L);
  }
  // 4) flash attention
  {
    int waves = B * NH * (L >> 4);
    flash_attn_kernel<<<waves / 8, 256, 0, stream>>>(q_h, k_h, v_t, a_h, L);
  }
  // 5) projection: [M,1024] @ [1024,1024]^T + bias -> f32 out
  {
    int waves = (M >> 5) * (DIMV >> 6);
    gemm_nt_wmma<<<(waves * 32) / 256, 256, 0, stream>>>(
        a_h, wp_h, proj_b, out, M, DIMV, DIMV);
  }
}
